// MultiheadAttention_2379411882045
// MI455X (gfx1250) — compile-verified, matrix operand order checked by reading, not run on hardware
//
#include <hip/hip_runtime.h>
#include <hip/hip_bf16.h>
#include <math.h>

typedef __attribute__((ext_vector_type(16))) __bf16 v16bf;
typedef __attribute__((ext_vector_type(8)))  __bf16 v8bf;
typedef __attribute__((ext_vector_type(8)))  float  v8f;

#define WMMA_BF16(A, B, C) \
  __builtin_amdgcn_wmma_f32_16x16x32_bf16(false, (A), false, (B), (short)0, (C), false, false)

// ---------------------------------------------------------------------------
// Fragment loader for 16-bit operand tiles with contiguous K in memory.
// A-matrix 16x32 layout (ISA 7.12.2): lanes 0-15 -> rows M=0..15 with K{0..7,16..23},
// lanes 16-31 -> rows M=0..15 with K{8..15,24..31}. B 32x16 mirrors with n=lane&15.
// row = (lane&15) row index within tile; half = lane>>4.
// ---------------------------------------------------------------------------
__device__ __forceinline__ v16bf load_frag16(const __bf16* __restrict__ base,
                                             int stride, int row, int k0, int half) {
  const __bf16* p = base + (size_t)row * stride + k0 + half * 8;
  v8bf lo = *(const v8bf*)(p);
  v8bf hi = *(const v8bf*)(p + 16);
  v16bf f;
#pragma unroll
  for (int i = 0; i < 8; i++) { f[i] = lo[i]; f[i + 8] = hi[i]; }
  return f;
}

// ---------------------------------------------------------------------------
// fp32 -> bf16 convert
// ---------------------------------------------------------------------------
__global__ void cvt_f32_bf16(const float* __restrict__ src, __bf16* __restrict__ dst, int n) {
  int i = blockIdx.x * blockDim.x + threadIdx.x;
  if (i < n) dst[i] = (__bf16)src[i];
}

// ---------------------------------------------------------------------------
// C[M=4096, N=1024] = A[4096,1024] * W[1024,1024]^T   (C[m,n] = sum_k A[m,k]W[n,k])
// MODE 0: store bf16 into (b,h,s,d) layout   (q/k projections)
// MODE 1: store fp32 row-major                (output projection)
// MODE 2: store bf16 into (b,h,d,s) layout    (V projection, pre-transposed
//         so attn@V B-fragments have contiguous K(=key) in memory)
// Block: 128 threads (4 waves). Wave computes a 16x64 tile: 4 accumulators,
// A-fragment reused across 4 B-fragments per K-step of 32.
// ---------------------------------------------------------------------------
template <int MODE>
__global__ __launch_bounds__(128) void gemm_bf16_wmma(const __bf16* __restrict__ A,
                                                      const __bf16* __restrict__ W,
                                                      void* __restrict__ out) {
  const int lane = threadIdx.x & 31;
  const int wave = threadIdx.x >> 5;
  const int half = lane >> 4;
  const int l15  = lane & 15;
  const int mBase = (blockIdx.y * 4 + wave) * 16;
  const int nBase = blockIdx.x * 64;

  v8f acc[4] = {};
  for (int k0 = 0; k0 < 1024; k0 += 32) {
    v16bf af = load_frag16(A, 1024, mBase + l15, k0, half);
    // keep the cache ahead of the K loop (lowers to global_prefetch_b8)
    if (k0 + 128 < 1024) {
      __builtin_prefetch(A + (size_t)(mBase + l15) * 1024 + k0 + 128, 0, 3);
      __builtin_prefetch(W + (size_t)(nBase + l15) * 1024 + k0 + 128, 0, 3);
    }
#pragma unroll
    for (int j = 0; j < 4; j++) {
      v16bf bf = load_frag16(W, 1024, nBase + j * 16 + l15, k0, half);
      acc[j] = WMMA_BF16(af, bf, acc[j]);
    }
  }

#pragma unroll
  for (int j = 0; j < 4; j++) {
#pragma unroll
    for (int r = 0; r < 8; r++) {
      const int m = mBase + r + half * 8;      // C: VGPR r, lanes<16 -> M=r, lanes>=16 -> M=r+8
      const int n = nBase + j * 16 + l15;      // N = lane&15
      const float v = acc[j][r];
      const int b = m >> 11, s = m & 2047;
      const int h = n >> 6,  d = n & 63;
      if (MODE == 0) {
        ((__bf16*)out)[(((size_t)(b * 16 + h) * 2048 + s) << 6) + d] = (__bf16)v;
      } else if (MODE == 2) {
        ((__bf16*)out)[(((size_t)(b * 16 + h) * 64 + d) << 11) + s] = (__bf16)v;
      } else {
        ((float*)out)[(size_t)m * 1024 + n] = v;
      }
    }
  }
}

// ---------------------------------------------------------------------------
// Attention: one workgroup = 16 query rows (strip) x all 2048 keys for one (b,h).
// 8 waves; wave w owns keys [w*256, w*256+256) = 16 key tiles.
// Scores strip lives entirely in registers (v8f sacc[16] per wave).
// vt is V pre-transposed to (b,h,d,s) so B-fragments load with vector ops.
// ---------------------------------------------------------------------------
__global__ __launch_bounds__(256) void attn_fused(const __bf16* __restrict__ qw,
                                                  const __bf16* __restrict__ kw,
                                                  const __bf16* __restrict__ vt,
                                                  const float* __restrict__ prev,
                                                  float* __restrict__ scores_out,
                                                  float* __restrict__ attn_out,
                                                  __bf16* __restrict__ ctx_out) {
  constexpr int S = 2048, DK = 64;
  const float scale = 0.125f;                  // 1/sqrt(64)
  const int strip = blockIdx.x;                // 0..127
  const int bh    = blockIdx.y;                // 0..31  (b*16+h)
  const int b     = bh >> 4;
  const int h     = bh & 15;
  const int lane  = threadIdx.x & 31;
  const int wave  = threadIdx.x >> 5;
  const int half  = lane >> 4;
  const int l15   = lane & 15;
  const size_t bhS = (size_t)bh * S;

  __shared__ float  redM[8][16];
  __shared__ float  redS[8][16];
  __shared__ float  rowMax[16];
  __shared__ float  rowInv[16];
  __shared__ float  ctxAcc[16 * 64];
  __shared__ __bf16 aStage[8][16 * 32];        // per-wave attn tile, A-frag staging

  for (int i = threadIdx.x; i < 16 * 64; i += 256) ctxAcc[i] = 0.0f;

  // Q strip A-fragments (16 rows x 64 d), loaded once per wave.
  const __bf16* qbase = qw + (bhS + (size_t)strip * 16) * DK;
  v16bf aq0 = load_frag16(qbase, DK, l15, 0, half);
  v16bf aq1 = load_frag16(qbase, DK, l15, 32, half);

  const int colBase = wave * 256;
  v8f   sacc[16];
  float maxr[8];
#pragma unroll
  for (int r = 0; r < 8; r++) maxr[r] = -__builtin_inff();

  // ---- Phase 1: scores = q k^T * scale + prev; write scores; track row max ----
#pragma unroll
  for (int t = 0; t < 16; t++) {
    const int col0 = colBase + t * 16;
    const __bf16* kbase = kw + (bhS + col0) * DK;
    v16bf kb0 = load_frag16(kbase, DK, l15, 0, half);   // B[k=d][n=key] : lane n = key row
    v16bf kb1 = load_frag16(kbase, DK, l15, 32, half);
    v8f acc = {};
    acc = WMMA_BF16(aq0, kb0, acc);
    acc = WMMA_BF16(aq1, kb1, acc);
#pragma unroll
    for (int r = 0; r < 8; r++) {
      const int row = strip * 16 + r + half * 8;
      const size_t idx = (bhS + row) * S + col0 + l15;
      const float s = acc[r] * scale + prev[idx];
      scores_out[idx] = s;
      maxr[r] = fmaxf(maxr[r], s);
      sacc[t][r] = s;
    }
  }

  // row-max: lanes 0..15 share rows 0..7, lanes 16..31 share rows 8..15
#pragma unroll
  for (int r = 0; r < 8; r++) {
    float m = maxr[r];
#pragma unroll
    for (int off = 1; off < 16; off <<= 1) m = fmaxf(m, __shfl_xor(m, off, 32));
    maxr[r] = m;
  }
  if (l15 == 0) {
#pragma unroll
    for (int r = 0; r < 8; r++) redM[wave][half * 8 + r] = maxr[r];
  }
  __syncthreads();
  if (threadIdx.x < 16) {
    float m = redM[0][threadIdx.x];
#pragma unroll
    for (int w = 1; w < 8; w++) m = fmaxf(m, redM[w][threadIdx.x]);
    rowMax[threadIdx.x] = m;
  }
  __syncthreads();

  // ---- Phase 2: exponentiate in-register, accumulate row sums ----
  float sumr[8];
#pragma unroll
  for (int r = 0; r < 8; r++) sumr[r] = 0.0f;
#pragma unroll
  for (int t = 0; t < 16; t++) {
#pragma unroll
    for (int r = 0; r < 8; r++) {
      const float p = __expf(sacc[t][r] - rowMax[half * 8 + r]);
      sacc[t][r] = p;
      sumr[r] += p;
    }
  }
#pragma unroll
  for (int r = 0; r < 8; r++) {
    float sm = sumr[r];
#pragma unroll
    for (int off = 1; off < 16; off <<= 1) sm += __shfl_xor(sm, off, 32);
    sumr[r] = sm;
  }
  if (l15 == 0) {
#pragma unroll
    for (int r = 0; r < 8; r++) redS[wave][half * 8 + r] = sumr[r];
  }
  __syncthreads();
  if (threadIdx.x < 16) {
    float sm = redS[0][threadIdx.x];
#pragma unroll
    for (int w = 1; w < 8; w++) sm += redS[w][threadIdx.x];
    rowInv[threadIdx.x] = 1.0f / sm;
  }
  __syncthreads();

  // ---- Phase 3: write attn, and context += attn @ V via WMMA ----
  const __bf16* vtb = vt + (((size_t)bh * 64) << 11);   // (d, s) plane for this head
  v8f ctx[4] = {};
#pragma unroll
  for (int c = 0; c < 8; c++) {                 // K-chunks of 32 keys
#pragma unroll
    for (int j = 0; j < 2; j++) {
      const int t = c * 2 + j;
      const int col0 = colBase + t * 16;
#pragma unroll
      for (int r = 0; r < 8; r++) {
        const int rloc = r + half * 8;
        const int row  = strip * 16 + rloc;
        const float a  = sacc[t][r] * rowInv[rloc];
        attn_out[(bhS + row) * S + col0 + l15] = a;
        aStage[wave][rloc * 32 + j * 16 + l15] = (__bf16)a;   // C-layout -> LDS
      }
    }
    asm volatile("s_wait_dscnt 0" ::: "memory");              // same-wave LDS RAW
    // rebuild attn tile as a 16x32 A-fragment from LDS
    v16bf af;
    {
      const __bf16* sp = &aStage[wave][l15 * 32 + half * 8];
#pragma unroll
      for (int i = 0; i < 8; i++) { af[i] = sp[i]; af[i + 8] = sp[i + 16]; }
    }
    // B frags from V^T: lane n = d (l15), K = keys, contiguous vector loads
#pragma unroll
    for (int nt = 0; nt < 4; nt++) {
      v16bf bv = load_frag16(vtb + ((size_t)(nt * 16) << 11), S, l15,
                             colBase + c * 32, half);
      ctx[nt] = WMMA_BF16(af, bv, ctx[nt]);
    }
  }

  // cross-wave context reduction in LDS
#pragma unroll
  for (int nt = 0; nt < 4; nt++) {
#pragma unroll
    for (int r = 0; r < 8; r++) {
      atomicAdd(&ctxAcc[(r + half * 8) * 64 + nt * 16 + l15], ctx[nt][r]);
    }
  }
  __syncthreads();

  // context -> (b, s, h*64+d) bf16 for the output projection GEMM
  for (int i = threadIdx.x; i < 16 * 64; i += 256) {
    const int rloc = i >> 6, d = i & 63;
    const int s = strip * 16 + rloc;
    ctx_out[((size_t)b * S + s) * 1024 + h * 64 + d] = (__bf16)ctxAcc[i];
  }
}

// ---------------------------------------------------------------------------
// Host side
// ---------------------------------------------------------------------------
extern "C" void kernel_launch(void* const* d_in, const int* in_sizes, int n_in,
                              void* d_out, int out_size, void* d_ws, size_t ws_size,
                              hipStream_t stream) {
  (void)in_sizes; (void)n_in; (void)out_size; (void)ws_size;
  constexpr size_t NX = 4096ull * 1024ull;     // 4,194,304 elems (X / proj / ctx)
  constexpr size_t NW = 1024ull * 1024ull;     // 1,048,576 elems (weights)

  const float* Q    = (const float*)d_in[0];
  const float* K    = (const float*)d_in[1];
  const float* V    = (const float*)d_in[2];
  const float* prev = (const float*)d_in[3];
  const float* W_Q  = (const float*)d_in[4];
  const float* W_K  = (const float*)d_in[5];
  const float* W_V  = (const float*)d_in[6];
  const float* W_O  = (const float*)d_in[7];

  float* out    = (float*)d_out;               // (B,S,1024)
  float* attn   = out + NX;                    // (B,H,S,S)
  float* scores = attn + 2ull * 16 * 2048 * 2048;

  // workspace carve-up (bf16 elements), 64 MB total
  __bf16* ws  = (__bf16*)d_ws;
  __bf16* Xq  = ws;            __bf16* Xk  = Xq + NX;  __bf16* Xv  = Xk + NX;
  __bf16* Wqb = Xv + NX;       __bf16* Wkb = Wqb + NW;
  __bf16* Wvb = Wkb + NW;      __bf16* Wob = Wvb + NW;
  __bf16* qw  = Wob + NW;      __bf16* kw  = qw + NX;  __bf16* vtw = kw + NX;
  __bf16* ctx = vtw + NX;

  const int T = 256;
  cvt_f32_bf16<<<(int)((NX + T - 1) / T), T, 0, stream>>>(Q, Xq, (int)NX);
  cvt_f32_bf16<<<(int)((NX + T - 1) / T), T, 0, stream>>>(K, Xk, (int)NX);
  cvt_f32_bf16<<<(int)((NX + T - 1) / T), T, 0, stream>>>(V, Xv, (int)NX);
  cvt_f32_bf16<<<(int)((NW + T - 1) / T), T, 0, stream>>>(W_Q, Wqb, (int)NW);
  cvt_f32_bf16<<<(int)((NW + T - 1) / T), T, 0, stream>>>(W_K, Wkb, (int)NW);
  cvt_f32_bf16<<<(int)((NW + T - 1) / T), T, 0, stream>>>(W_V, Wvb, (int)NW);
  cvt_f32_bf16<<<(int)((NW + T - 1) / T), T, 0, stream>>>(W_O, Wob, (int)NW);

  dim3 gg(16, 64), gb(128);
  gemm_bf16_wmma<0><<<gg, gb, 0, stream>>>(Xq, Wqb, qw);
  gemm_bf16_wmma<0><<<gg, gb, 0, stream>>>(Xk, Wkb, kw);
  gemm_bf16_wmma<2><<<gg, gb, 0, stream>>>(Xv, Wvb, vtw);   // V stored transposed

  attn_fused<<<dim3(128, 32), 256, 0, stream>>>(qw, kw, vtw, prev, scores, attn, ctx);

  gemm_bf16_wmma<1><<<gg, gb, 0, stream>>>(ctx, Wob, out);
}
